// PopulationAttentionDispersed_59682865545946
// MI455X (gfx1250) — matrix operand order repaired and checked by reading, not compiled, not run on hardware
//
#include <hip/hip_runtime.h>

// ---------------------------------------------------------------------------
// PopulationAttentionDispersed for MI455X (gfx1250, wave32, WMMA + async LDS).
//
//   S = Q K^T / 8                         (per b,h; L=2048, d=64)
//   softmax over the *query* axis, weighted by x[b,q]:
//     m[k] = max_q S[q,k];  Z[k] = sum_q x[q] exp(S[q,k]-m[k])
//     beta[k] = m[k] - log(x[k]/Z[k])
//   O[q,:] = x[q] * sum_k exp(S[q,k]-beta[k]) * V[k,:]
//
// bf16 WMMA 16x16x32 with f32 accumulation; K/V (pass 2) and Q (pass 1)
// tiles staged workgroup-wide via GLOBAL_LOAD_ASYNC_TO_LDS_B128, double
// buffered so copy(i+1) overlaps compute(i). Lane-private online-softmax
// stats merged once after the stream (no per-iteration cross-lane traffic).
// ---------------------------------------------------------------------------

#define PA_B 2
#define PA_H 16
#define PA_L 2048
#define PA_D 64

typedef __attribute__((ext_vector_type(16))) __bf16 v16bf;
typedef __attribute__((ext_vector_type(8)))  float  v8f;

union BF16Op {           // one WMMA A/B operand: 32 lanes x 16 bf16 = 8 VGPRs
  v16bf v;
  uint4 q[2];
};

__device__ __forceinline__ unsigned short f32_to_bf16(float f) {
  union { float f; unsigned int u; } v; v.f = f;
  unsigned int u = v.u;
  u += 0x7FFFu + ((u >> 16) & 1u);   // round-to-nearest-even
  return (unsigned short)(u >> 16);
}

__device__ __forceinline__ v8f wmma_bf16(const BF16Op& a, const BF16Op& b, v8f c) {
  return __builtin_amdgcn_wmma_f32_16x16x32_bf16(false, a.v, false, b.v,
                                                 (short)0, c, false, false);
}

// Async DMA: 16B global -> LDS, tracked by ASYNCcnt (CDNA5 §10.7/§15.18).
// VDST VGPR carries the LDS byte offset; VADDR is the 64-bit global address.
__device__ __forceinline__ void async_copy_b128(unsigned int lds_off,
                                                const void* gptr) {
  unsigned long long ga = (unsigned long long)(size_t)gptr;
  asm volatile("global_load_async_to_lds_b128 %0, %1, off"
               :: "v"(lds_off), "v"(ga) : "memory");
}
__device__ __forceinline__ void wait_async0() {
  asm volatile("s_wait_asynccnt 0x0" ::: "memory");
}
__device__ __forceinline__ unsigned int lds_off_of(const void* p) {
  return (unsigned int)(size_t)p;   // low 32 bits of flat addr = LDS offset
}

// --------------------------- prep: f32 -> bf16 ------------------------------
__global__ __launch_bounds__(256) void pa_convert(const float* __restrict__ Q,
                                                  const float* __restrict__ K,
                                                  unsigned short* __restrict__ Qh,
                                                  unsigned short* __restrict__ Kh,
                                                  int N) {
  int i = blockIdx.x * 256 + threadIdx.x;
  if (i < N) {
    Qh[i] = f32_to_bf16(Q[i] * 0.125f);   // fold 1/sqrt(64), exact
    Kh[i] = f32_to_bf16(K[i]);
  }
}

// ----------------------- prep: V -> bf16, transposed ------------------------
// Vt[bh][d][k] = bf16(V[bh][k][d]) so PV B-operands are contiguous along k.
__global__ __launch_bounds__(256) void pa_transpose_v(const float* __restrict__ V,
                                                      unsigned short* __restrict__ Vt) {
  __shared__ unsigned short tile[16][17];
  const int tx = threadIdx.x, ty = threadIdx.y;
  const int bh = blockIdx.z;
  const int k0 = blockIdx.x * 16;
  const int d0 = blockIdx.y * 16;
  tile[ty][tx] = f32_to_bf16(V[((size_t)bh * PA_L + (k0 + ty)) * PA_D + d0 + tx]);
  __syncthreads();
  Vt[((size_t)bh * PA_D + (d0 + ty)) * PA_L + k0 + tx] = tile[tx][ty];
}

// ------------------- pass 1: per-key column stats -> beta -------------------
// 4 waves/WG own 64 key columns; the shared Q stream is staged into LDS with
// async copies (one 2KB tile per step, double buffered).
__global__ __launch_bounds__(128) void pa_pass1(const unsigned short* __restrict__ Qh,
                                                const unsigned short* __restrict__ Kh,
                                                const float* __restrict__ x,
                                                float* __restrict__ beta) {
  __shared__ __align__(16) unsigned short sQ[2][16][64];   // 2 x 2KB
  const int tid  = threadIdx.x;
  const int lane = tid & 31;
  const int wid  = tid >> 5;
  const int bh   = blockIdx.y;
  const int b    = bh >> 4;                     // H = 16
  const int kb   = (blockIdx.x * 4 + wid) * 16;
  const int n    = lane & 15;
  const int hi   = lane >> 4;

  // B operands (K^T, contraction = d), fixed for this wave's 16 keys.
  const unsigned short* Krow = Kh + ((size_t)bh * PA_L + (kb + n)) * PA_D;
  BF16Op bK0, bK1;
  bK0.q[0] = *(const uint4*)(Krow + 16 * hi);
  bK0.q[1] = *(const uint4*)(Krow + 16 * hi + 8);
  bK1.q[0] = *(const uint4*)(Krow + 32 + 16 * hi);
  bK1.q[1] = *(const uint4*)(Krow + 32 + 16 * hi + 8);

  const float* xb = x + b * PA_L;
  const unsigned short* Qbase = Qh + (size_t)bh * PA_L * PA_D;
  const unsigned int ldsQ[2] = { lds_off_of(&sQ[0][0][0]), lds_off_of(&sQ[1][0][0]) };

  // Prologue: stage q-tile 0 (2KB contiguous = 128 x 16B, one chunk/thread).
  async_copy_b128(ldsQ[0] + tid * 16, Qbase + tid * 8);

  float m_run = -3.0e38f;     // lane-private stats over rows q = 16*i + r + 8*hi
  float Z = 0.0f;

  for (int i = 0; i < PA_L / 16; ++i) {
    const int buf = i & 1;
    wait_async0();            // my chunk of tile i has landed
    __syncthreads();          // everyone's chunks landed; buf^1 reads retired
    if (i + 1 < PA_L / 16)    // overlap: stage tile i+1 while computing tile i
      async_copy_b128(ldsQ[buf ^ 1] + tid * 16,
                      Qbase + (size_t)(i + 1) * 16 * PA_D + tid * 8);

    const unsigned short* Qrow = &sQ[buf][n][0];
    BF16Op aQ0, aQ1;
    aQ0.q[0] = *(const uint4*)(Qrow + 16 * hi);
    aQ0.q[1] = *(const uint4*)(Qrow + 16 * hi + 8);
    aQ1.q[0] = *(const uint4*)(Qrow + 32 + 16 * hi);
    aQ1.q[1] = *(const uint4*)(Qrow + 32 + 16 * hi + 8);

    v8f s = {};
    s = wmma_bf16(aQ0, bK0, s);
    s = wmma_bf16(aQ1, bK1, s);

    float xq[8];
    {
      const float4* xp = (const float4*)(xb + i * 16 + 8 * hi);
      float4 x0 = xp[0], x1 = xp[1];
      xq[0] = x0.x; xq[1] = x0.y; xq[2] = x0.z; xq[3] = x0.w;
      xq[4] = x1.x; xq[5] = x1.y; xq[6] = x1.z; xq[7] = x1.w;
    }

    float tm = -3.0e38f;
#pragma unroll
    for (int r = 0; r < 8; ++r) tm = fmaxf(tm, s[r]);
    const float m_new = fmaxf(m_run, tm);
    float ps = 0.0f;
#pragma unroll
    for (int r = 0; r < 8; ++r) ps += xq[r] * __expf(s[r] - m_new);
    Z = Z * __expf(m_run - m_new) + ps;
    m_run = m_new;
  }

  // One-shot merge of the two half-columns (lane n <-> lane n+16).
  const float om = __shfl_xor(m_run, 16, 32);
  const float oZ = __shfl_xor(Z, 16, 32);
  const float m  = fmaxf(m_run, om);
  const float Zc = Z * __expf(m_run - m) + oZ * __expf(om - m);

  if (hi == 0) {
    const int k = kb + n;
    // beta[k] = m[k] - log(x[k]) + log(Z[k]);  exp(S-beta) = exp(S-m)*x_k/Z
    beta[(size_t)bh * PA_L + k] = m - __logf(xb[k]) + __logf(Zc);
  }
}

// ------------------- pass 2: O = x_q * exp(S - beta) @ V --------------------
// 4 waves/WG own 64 q-rows; each 32-key K/V block (8KB) is staged once per WG
// via async copies (double buffered), cutting per-wave L2 traffic 4x.
__global__ __launch_bounds__(128) void pa_pass2(const unsigned short* __restrict__ Qh,
                                                const unsigned short* __restrict__ Kh,
                                                const unsigned short* __restrict__ Vt,
                                                const float* __restrict__ x,
                                                const float* __restrict__ beta,
                                                float* __restrict__ O) {
  __shared__ __align__(16) unsigned short sK[2][32][64];   // 2 x 4KB
  __shared__ __align__(16) unsigned short sV[2][64][32];   // 2 x 4KB
  __shared__ __align__(16) unsigned short sP[4][16][40];   // per-wave C->A relayout
  const int tid  = threadIdx.x;
  const int lane = tid & 31;
  const int wid  = tid >> 5;
  const int bh   = blockIdx.y;
  const int b    = bh >> 4;
  const int qb   = (blockIdx.x * 4 + wid) * 16;
  const int n    = lane & 15;
  const int hi   = lane >> 4;

  // A operands for this wave's 16 Q rows (live in VGPRs for the whole sweep).
  const unsigned short* Qrow = Qh + ((size_t)bh * PA_L + (qb + n)) * PA_D;
  BF16Op aQ0, aQ1;
  aQ0.q[0] = *(const uint4*)(Qrow + 16 * hi);
  aQ0.q[1] = *(const uint4*)(Qrow + 16 * hi + 8);
  aQ1.q[0] = *(const uint4*)(Qrow + 32 + 16 * hi);
  aQ1.q[1] = *(const uint4*)(Qrow + 32 + 16 * hi + 8);

  const unsigned short* Kbase = Kh + (size_t)bh * PA_L * PA_D;
  const unsigned short* Vbase = Vt + (size_t)bh * PA_D * PA_L;
  const float* betab = beta + (size_t)bh * PA_L;
  const unsigned int ldsK[2] = { lds_off_of(&sK[0][0][0]), lds_off_of(&sK[1][0][0]) };
  const unsigned int ldsV[2] = { lds_off_of(&sV[0][0][0]), lds_off_of(&sV[1][0][0]) };

  // Stage one 32-key block (K: 4KB contiguous; V: 64 rows x 64B), 4 chunks/thread.
  auto issue_kv = [&](int blk, int buf) {
    const unsigned short* Kg = Kbase + (size_t)blk * 32 * PA_D;  // contiguous 4KB
    async_copy_b128(ldsK[buf] + tid * 16,        Kg + tid * 8);
    async_copy_b128(ldsK[buf] + tid * 16 + 2048, Kg + (tid + 128) * 8);
    const unsigned short* Vg = Vbase + blk * 32;
    const int c0 = tid, c1 = tid + 128;          // chunk c -> row d=c>>2, 16B piece c&3
    async_copy_b128(ldsV[buf] + c0 * 16, Vg + (size_t)(c0 >> 2) * PA_L + (c0 & 3) * 8);
    async_copy_b128(ldsV[buf] + c1 * 16, Vg + (size_t)(c1 >> 2) * PA_L + (c1 & 3) * 8);
  };

  v8f acc0 = {}, acc1 = {}, acc2 = {}, acc3 = {};
  unsigned short (*myP)[40] = sP[wid];

  issue_kv(0, 0);                                 // prologue
  const int NBLK = PA_L / 32;                     // 64
  for (int i = 0; i < NBLK; ++i) {
    const int buf = i & 1;
    wait_async0();                                // block i landed (my 4 chunks)
    __syncthreads();                              // all chunks in; buf^1 reads done
    if (i + 1 < NBLK) issue_kv(i + 1, buf ^ 1);   // overlap copy(i+1) w/ compute(i)

#pragma unroll
    for (int t = 0; t < 2; ++t) {
      const unsigned short* Kr = &sK[buf][t * 16 + n][0];
      BF16Op bK0, bK1;
      bK0.q[0] = *(const uint4*)(Kr + 16 * hi);
      bK0.q[1] = *(const uint4*)(Kr + 16 * hi + 8);
      bK1.q[0] = *(const uint4*)(Kr + 32 + 16 * hi);
      bK1.q[1] = *(const uint4*)(Kr + 32 + 16 * hi + 8);

      v8f s = {};
      s = wmma_bf16(aQ0, bK0, s);
      s = wmma_bf16(aQ1, bK1, s);

      const float bk = betab[i * 32 + t * 16 + n];   // column bias, k = 32i+16t+n
#pragma unroll
      for (int r = 0; r < 8; ++r) {
        const float p = __expf(s[r] - bk);           // = exp(S-m) * x_k / Z_k
        myP[r + 8 * hi][t * 16 + n] = f32_to_bf16(p);
      }
    }

    // Re-read the 16x32 P tile in WMMA A layout (per-wave LDS, DScnt-ordered).
    BF16Op aP;
    {
      const unsigned short* prow = &myP[n][0];
      aP.q[0] = *(const uint4*)(prow + 16 * hi);
      aP.q[1] = *(const uint4*)(prow + 16 * hi + 8);
    }

    // O(16x64) += P(16x32) @ V(32x64) from the shared, pre-transposed V block.
#pragma unroll
    for (int dt = 0; dt < 4; ++dt) {
      const unsigned short* Vr = &sV[buf][dt * 16 + n][0] + 16 * hi;
      BF16Op bV;
      bV.q[0] = *(const uint4*)(Vr);
      bV.q[1] = *(const uint4*)(Vr + 8);
      if      (dt == 0) acc0 = wmma_bf16(aP, bV, acc0);
      else if (dt == 1) acc1 = wmma_bf16(aP, bV, acc1);
      else if (dt == 2) acc2 = wmma_bf16(aP, bV, acc2);
      else              acc3 = wmma_bf16(aP, bV, acc3);
    }
  }

  // Epilogue: scale rows by x[b,q] and store (lane covers q = qb + r + 8*hi).
  float xq[8];
  {
    const float4* xp = (const float4*)(x + b * PA_L + qb + 8 * hi);
    float4 x0 = xp[0], x1 = xp[1];
    xq[0] = x0.x; xq[1] = x0.y; xq[2] = x0.z; xq[3] = x0.w;
    xq[4] = x1.x; xq[5] = x1.y; xq[6] = x1.z; xq[7] = x1.w;
  }
#pragma unroll
  for (int r = 0; r < 8; ++r) {
    float* orow = O + ((size_t)bh * PA_L + (qb + r + 8 * hi)) * PA_D + n;
    orow[0]  = acc0[r] * xq[r];
    orow[16] = acc1[r] * xq[r];
    orow[32] = acc2[r] * xq[r];
    orow[48] = acc3[r] * xq[r];
  }
}

// ---------------------------------------------------------------------------
extern "C" void kernel_launch(void* const* d_in, const int* in_sizes, int n_in,
                              void* d_out, int out_size, void* d_ws, size_t ws_size,
                              hipStream_t stream) {
  (void)in_sizes; (void)n_in; (void)out_size; (void)ws_size;
  const float* Q = (const float*)d_in[0];
  const float* K = (const float*)d_in[1];
  const float* V = (const float*)d_in[2];
  const float* x = (const float*)d_in[3];
  float* O = (float*)d_out;

  const int BH = PA_B * PA_H;
  const int N  = BH * PA_L * PA_D;              // 4,194,304 elements
  const size_t SZH = (size_t)N * 2;             // 8 MiB per bf16 tensor

  char* ws = (char*)d_ws;
  unsigned short* Qh  = (unsigned short*)(ws);
  unsigned short* Kh  = (unsigned short*)(ws + SZH);
  unsigned short* Vt  = (unsigned short*)(ws + 2 * SZH);
  float*          bet = (float*)(ws + 3 * SZH);

  pa_convert<<<N / 256, 256, 0, stream>>>(Q, K, Qh, Kh, N);
  pa_transpose_v<<<dim3(PA_L / 16, PA_D / 16, BH), dim3(16, 16), 0, stream>>>(V, Vt);
  pa_pass1<<<dim3(PA_L / 64, BH), 128, 0, stream>>>(Qh, Kh, x, bet);
  pa_pass2<<<dim3(PA_L / 64, BH), 128, 0, stream>>>(Qh, Kh, Vt, x, bet, O);
}
